// Double_Attention_Model_2259152797766
// MI455X (gfx1250) — matrix operand order repaired and checked by reading, not compiled, not run on hardware
//
#include <hip/hip_runtime.h>

typedef __attribute__((ext_vector_type(16))) __bf16 v16bf;
typedef __attribute__((ext_vector_type(8)))  __bf16 v8bf;
typedef __attribute__((ext_vector_type(8)))  float  v8f;
typedef __attribute__((ext_vector_type(4)))  unsigned int u32x4;
typedef __attribute__((ext_vector_type(8)))  unsigned int u32x8;

// Load 8 contiguous elements at p, convert to bf16 into dst[off..off+7].
__device__ __forceinline__ void cvt8(v16bf& dst, int off, const float* p) {
    v8f t = *(const v8f*)p;
#pragma unroll
    for (int e = 0; e < 8; ++e) dst[off + e] = (__bf16)t[e];
}
__device__ __forceinline__ void cvt8(v16bf& dst, int off, const __bf16* p) {
    v8bf t = *(const v8bf*)p;
#pragma unroll
    for (int e = 0; e < 8; ++e) dst[off + e] = t[e];
}

// ---------------------------------------------------------------------------
// TDM: DMA a 2D tile (rows x 64 elems, row-major, contiguous stride 64) from
// global memory into LDS. D# groups per CDNA5 ISA.
// 2-group form (tensors <= 2D): VADDR2/VADDR3 disabled.
// ---------------------------------------------------------------------------
__device__ __forceinline__ void tdm_load_tile_2d(unsigned int lds_off,
                                                 unsigned long long gaddr,
                                                 unsigned int data_size_code, // 1=2B, 2=4B
                                                 unsigned int rows,
                                                 unsigned long long n_rows_total)
{
    u32x4 g0;
    g0[0] = 1u;                                               // count=1, user mode
    g0[1] = lds_off;                                          // lds_addr (bytes)
    g0[2] = (unsigned int)(gaddr & 0xffffffffull);            // global_addr[31:0]
    g0[3] = (unsigned int)((gaddr >> 32) & 0x1ffffffull)      // global_addr[56:32]
            | 0x80000000u;                                    // type=2 ("image")

    u32x8 g1;
    g1[0] = data_size_code << 16;                             // workgroup_mask=0 | data_size
    g1[1] = 64u << 16;                                        // tensor_dim0 = 64 (lo16)
    g1[2] = ((unsigned int)(n_rows_total & 0xffffull)) << 16; // tensor_dim1 lo16
    g1[3] = (unsigned int)((n_rows_total >> 16) & 0xffffull)  // tensor_dim1 hi16
            | (64u << 16);                                    // tile_dim0 = 64
    g1[4] = rows & 0xffffu;                                   // tile_dim1 | tile_dim2=0
    g1[5] = 64u;                                              // tensor_dim0_stride = 64
    g1[6] = 0u;
    g1[7] = 0u;

    asm volatile("tensor_load_to_lds %0, %1" :: "s"(g0), "s"(g1) : "memory");
}

// ---------------------------------------------------------------------------
// Generic Y(M,64) = X(M,64) @ W(64,64)^T   (Y[m,n] = sum_e X[m,e] * W[n,e])
// One wave per 128-row strip: W fragments loaded once, 8 m-tiles x 8 WMMA = 64
// v_wmma_f32_16x16x32_bf16 per wave. USE_TDM: A tile staged via TDM into LDS,
// double-buffered in two 64-row halves overlapping DMA with WMMA.
// ---------------------------------------------------------------------------
template <typename InT, typename OutT, bool USE_TDM>
__global__ __launch_bounds__(32)
void gemm64_wmma(const InT* __restrict__ X, const float* __restrict__ W,
                 OutT* __restrict__ Y,
                 long long block_rows, long long block_stride, long long row_stride,
                 long long M)
{
    extern __shared__ char smem_raw[];
    InT* tileA = (InT*)smem_raw;

    const int lane = threadIdx.x;
    const long long m0 = (long long)blockIdx.x * 128;
    const int rrow = lane & 15;
    const int half = lane >> 4;
    const int koff = half * 8;

    // ---- B fragments: B[k][n] = W[n][k]; lane holds column n = nt*16+rrow,
    //      K = kc*32 + half*16 .. +15 (contiguous within a W row).
    v16bf Bf[4][2];
#pragma unroll
    for (int nt = 0; nt < 4; ++nt) {
#pragma unroll
        for (int kc = 0; kc < 2; ++kc) {
            const float* wrow = W + (nt * 16 + rrow) * 64 + kc * 32 + half * 16;
            cvt8(Bf[nt][kc], 0, wrow);
            cvt8(Bf[nt][kc], 8, wrow + 8);
        }
    }

    if constexpr (USE_TDM) {
        const unsigned long long base = (unsigned long long)(uintptr_t)(X + m0 * 64);
        const unsigned int esz = (unsigned int)sizeof(InT);
        const unsigned int dsc = (esz == 2u) ? 1u : 2u;
        const unsigned int lds0 = (unsigned int)(uintptr_t)tileA; // flat LDS addr[31:0]
        tdm_load_tile_2d(lds0,               base,                   dsc, 64, (unsigned long long)M);
        tdm_load_tile_2d(lds0 + 64u*64u*esz, base + 64ull*64ull*esz, dsc, 64, (unsigned long long)M);
    }

#pragma unroll
    for (int mt = 0; mt < 8; ++mt) {
        if constexpr (USE_TDM) {
            if (mt == 0) { __builtin_amdgcn_s_wait_tensorcnt(1); asm volatile("" ::: "memory"); }
            if (mt == 4) { __builtin_amdgcn_s_wait_tensorcnt(0); asm volatile("" ::: "memory"); }
        }

        const InT* arow;
        if constexpr (USE_TDM) {
            arow = tileA + (mt * 16 + rrow) * 64;
        } else {
            const long long m = m0 + mt * 16 + rrow;
            arow = X + (m / block_rows) * block_stride + (m % block_rows) * row_stride;
        }
        v16bf Af[2];
#pragma unroll
        for (int kc = 0; kc < 2; ++kc) {
            const InT* xp = arow + kc * 32 + half * 16;
            cvt8(Af[kc], 0, xp);
            cvt8(Af[kc], 8, xp + 8);
        }

        v8f acc[4] = {};
#pragma unroll
        for (int nt = 0; nt < 4; ++nt) {
            acc[nt] = __builtin_amdgcn_wmma_f32_16x16x32_bf16(
                false, Af[0], false, Bf[nt][0], (short)0, acc[nt], false, false);
            acc[nt] = __builtin_amdgcn_wmma_f32_16x16x32_bf16(
                false, Af[1], false, Bf[nt][1], (short)0, acc[nt], false, false);
        }

#pragma unroll
        for (int nt = 0; nt < 4; ++nt) {
#pragma unroll
            for (int r = 0; r < 8; ++r) {
                long long row = m0 + mt * 16 + r + half * 8;
                Y[row * 64 + nt * 16 + rrow] = (OutT)acc[nt][r];
            }
        }
    }
}

// ---------------------------------------------------------------------------
// Stage-1 attention, TRANSPOSED formulation: one wave per (b,h,j-group-of-4).
//   S^T[s',i] = sum_d k[b,s',j,h,d] * q[b,i,h,d]    (32x64, K=32 -> 8 WMMA)
//   softmax over s' = per-COLUMN: 16 in-lane values + one xor-16 exchange
//   T^T[d,i]  = sum_s' v[b,s',j,h,d] * P[s',i]      (32x64, K=32 -> 8 WMMA)
// q: (B,N,64) bf16; k,v: (B,S,N,64) bf16; Tpre: (B,N,N,64) bf16
// ---------------------------------------------------------------------------
__global__ __launch_bounds__(32)
void stage1_attn(const __bf16* __restrict__ q, const __bf16* __restrict__ k,
                 const __bf16* __restrict__ v, __bf16* __restrict__ Tpre)
{
    const int bid = blockIdx.x;         // ((b*2 + h)*16 + jg)
    const int jg = bid & 15;
    const int h = (bid >> 4) & 1;
    const int b = bid >> 5;
    const int lane = threadIdx.x;
    const int rrow = lane & 15;
    const int half = lane >> 4;
    const int koff = half * 8;
    const float scale = 0.17677669529663687f; // 1/sqrt(32)

    __shared__ __bf16 smem1[64][32];    // probs^T [i][s'], then outT [i][d]

    // B fragments from q^T: column i = nt*16+rrow, K = d = half*16..+15
    // (single contiguous 32B load per tile; loaded once, reused over 4 j)
    v16bf Bq[4];
#pragma unroll
    for (int nt = 0; nt < 4; ++nt)
        Bq[nt] = *(const v16bf*)(q + ((long long)b * 64 + nt * 16 + rrow) * 64 + h * 32 + half * 16);

    for (int jj = 0; jj < 4; ++jj) {
        const int j = jg * 4 + jj;

        // A fragments from k: row s' = mt*16+rrow, K = d = half*16..+15
        v16bf Ak[2];
#pragma unroll
        for (int mt = 0; mt < 2; ++mt) {
            const __bf16* kr = k + (((long long)(b * 32 + mt * 16 + rrow) * 64 + j) * 64) + h * 32;
            cvt8(Ak[mt], 0, kr + half * 16);
            cvt8(Ak[mt], 8, kr + half * 16 + 8);
        }

        v8f st[2][4] = {};
#pragma unroll
        for (int mt = 0; mt < 2; ++mt)
#pragma unroll
            for (int nt = 0; nt < 4; ++nt)
                st[mt][nt] = __builtin_amdgcn_wmma_f32_16x16x32_bf16(
                    false, Ak[mt], false, Bq[nt], (short)0, st[mt][nt], false, false);

        // Column softmax: lane holds s' in {8*half + r + 16*mt}; lane^16 holds
        // the complementary 16. One exchange for max, one for sum, per nt.
#pragma unroll
        for (int nt = 0; nt < 4; ++nt) {
            float mx = -3.4e38f;
#pragma unroll
            for (int mt = 0; mt < 2; ++mt)
#pragma unroll
                for (int r = 0; r < 8; ++r) mx = fmaxf(mx, st[mt][nt][r] * scale);
            mx = fmaxf(mx, __shfl_xor(mx, 16, 32));
            float e[2][8];
            float sm = 0.f;
#pragma unroll
            for (int mt = 0; mt < 2; ++mt)
#pragma unroll
                for (int r = 0; r < 8; ++r) {
                    e[mt][r] = __expf(st[mt][nt][r] * scale - mx);
                    sm += e[mt][r];
                }
            sm += __shfl_xor(sm, 16, 32);
            const float inv = 1.0f / sm;
            const int i = nt * 16 + rrow;
#pragma unroll
            for (int mt = 0; mt < 2; ++mt) {
                v8bf pv;
#pragma unroll
                for (int r = 0; r < 8; ++r) pv[r] = (__bf16)(e[mt][r] * inv);
                *(v8bf*)&smem1[i][mt * 16 + 8 * half] = pv;  // s' = mt*16+8*half+r
            }
        }
        __syncthreads();

        // A fragments from v^T: row d = mt*16+rrow, K = s' (strided gather)
        v16bf Av[2];
#pragma unroll
        for (int mt = 0; mt < 2; ++mt) {
            const int d = mt * 16 + rrow;
            const __bf16* vb = v + ((long long)b * 32 * 64 + j) * 64 + h * 32 + d;
#pragma unroll
            for (int t = 0; t < 8; ++t) {
                Av[mt][t]     = vb[(long long)(half * 16 + t) * 4096];
                Av[mt][8 + t] = vb[(long long)(half * 16 + 8 + t) * 4096];
            }
        }
        // B fragments from probs^T: column i = nt*16+rrow, K = s' = half*16..+15
        v16bf Bp[4];
#pragma unroll
        for (int nt = 0; nt < 4; ++nt)
            Bp[nt] = *(const v16bf*)&smem1[nt * 16 + rrow][half * 16];
        __syncthreads();

        v8f tt[2][4] = {};
#pragma unroll
        for (int mt = 0; mt < 2; ++mt)
#pragma unroll
            for (int nt = 0; nt < 4; ++nt)
                tt[mt][nt] = __builtin_amdgcn_wmma_f32_16x16x32_bf16(
                    false, Av[mt], false, Bp[nt], (short)0, tt[mt][nt], false, false);

        // Bounce T^T tile through LDS as [i][d], then coalesced 32B row stores.
#pragma unroll
        for (int mt = 0; mt < 2; ++mt)
#pragma unroll
            for (int nt = 0; nt < 4; ++nt) {
                const int i = nt * 16 + rrow;
                v8bf ov;
#pragma unroll
                for (int r = 0; r < 8; ++r) ov[r] = (__bf16)tt[mt][nt][r];
                *(v8bf*)&smem1[i][mt * 16 + 8 * half] = ov;  // d = mt*16+8*half+r
            }
        __syncthreads();

#pragma unroll
        for (int rr = 0; rr < 2; ++rr) {
            const int i = lane + rr * 32;
            __bf16* dst = Tpre + (((long long)(b * 64 + i) * 64 + j) * 64) + h * 32;
            *(v16bf*)dst = *(const v16bf*)&smem1[i][0];
            *(v16bf*)(dst + 16) = *(const v16bf*)&smem1[i][16];
        }
        __syncthreads();
    }
}

// ---------------------------------------------------------------------------
// Stage-2 attention (batched GEMV; ~67 MFLOP total -> VALU + LDS).
// One 64-thread block per (b,i); thread t = j.
// ---------------------------------------------------------------------------
__global__ __launch_bounds__(64)
void stage2_attn(const __bf16* __restrict__ q2, const __bf16* __restrict__ k2,
                 const __bf16* __restrict__ v2, const int* __restrict__ adj,
                 __bf16* __restrict__ opre)
{
    const int bi = blockIdx.x;
    const int b = bi >> 6;
    const int i = bi & 63;
    const int j = threadIdx.x;
    const float scale = 0.17677669529663687f;

    __shared__ float sc[2][64];
    __shared__ float red[64][65];

    const long long base_bi = (long long)b * 64 + i;
    const __bf16* qrow = q2 + base_bi * 64;
    const __bf16* krow = k2 + (base_bi * 64 + j) * 64;
    const int masked = adj[i * 64 + j] > 0;

    float s2[2];
#pragma unroll
    for (int h = 0; h < 2; ++h) {
        float dot = 0.f;
#pragma unroll
        for (int d = 0; d < 32; ++d)
            dot += (float)qrow[h * 32 + d] * (float)krow[h * 32 + d];
        s2[h] = masked ? dot * scale : -1.0e9f;
        sc[h][j] = s2[h];
    }
    __syncthreads();

    float a2[2];
#pragma unroll
    for (int h = 0; h < 2; ++h) {
        float mx = -3.4e38f;
        for (int t = 0; t < 64; ++t) mx = fmaxf(mx, sc[h][t]);
        float sm = 0.f;
        for (int t = 0; t < 64; ++t) sm += __expf(sc[h][t] - mx);
        a2[h] = __expf(s2[h] - mx) / sm;
    }

    const __bf16* vrow = v2 + (base_bi * 64 + j) * 64;
#pragma unroll
    for (int c = 0; c < 64; ++c) red[j][c] = a2[c >> 5] * (float)vrow[c];
    __syncthreads();

    float accv = 0.f;
    for (int t = 0; t < 64; ++t) accv += red[t][j];
    opre[base_bi * 64 + j] = (__bf16)accv;
}

// ---------------------------------------------------------------------------
extern "C" void kernel_launch(void* const* d_in, const int* in_sizes, int n_in,
                              void* d_out, int out_size, void* d_ws, size_t ws_size,
                              hipStream_t stream)
{
    const float* x    = (const float*)d_in[0];
    const int*   adj  = (const int*)d_in[1];
    const float* Wq_t = (const float*)d_in[2];
    const float* Wk_t = (const float*)d_in[3];
    const float* Wv_t = (const float*)d_in[4];
    const float* Wo_t = (const float*)d_in[5];
    const float* Wq_s = (const float*)d_in[6];
    const float* Wk_s = (const float*)d_in[7];
    const float* Wv_s = (const float*)d_in[8];
    const float* Wo_s = (const float*)d_in[9];
    float* out = (float*)d_out;

    // B=128, S=32, N=64, E=64
    const long long M_kv = 128ll * 32 * 64;  // 262144
    const long long M_q  = 128ll * 64;       // 8192
    const long long M_T  = 128ll * 64 * 64;  // 524288

    char* ws = (char*)d_ws;
    __bf16* q    = (__bf16*)(ws + 0);                 // 1 MB
    __bf16* q2   = (__bf16*)(ws + (1ll << 20));       // 1 MB
    __bf16* opre = (__bf16*)(ws + (2ll << 20));       // 1 MB
    __bf16* kbuf = (__bf16*)(ws + (4ll << 20));       // 33.55 MB
    __bf16* vbuf = (__bf16*)(ws + (4ll << 20) + 33554432ll);
    __bf16* Tpre = (__bf16*)(ws + 71303168ll);        // 67.1 MB (region B)
    __bf16* Tm   = (__bf16*)(ws + 138412032ll);       // 67.1 MB (region C)
    __bf16* k2   = kbuf;                              // reuse region A after stage1
    __bf16* v2   = Tpre;                              // reuse region B after T GEMM

    const size_t smem_f32  = 128 * 64 * sizeof(float);   // 32 KB
    const size_t smem_bf16 = 128 * 64 * sizeof(__bf16);  // 16 KB

    // 1) projections from x (f32 -> bf16), A staged via TDM
    gemm64_wmma<float, __bf16, true><<<(int)(M_kv / 128), 32, smem_f32, stream>>>(
        x, Wk_t, kbuf, M_kv, 0, 64, M_kv);
    gemm64_wmma<float, __bf16, true><<<(int)(M_kv / 128), 32, smem_f32, stream>>>(
        x, Wv_t, vbuf, M_kv, 0, 64, M_kv);
    // curr = x[:, -1]: row (b*64+n) -> x + 31*64*64 + b*(32*64*64) + n*64
    const float* curr = x + 31ll * 64 * 64;
    gemm64_wmma<float, __bf16, false><<<(int)(M_q / 128), 32, 0, stream>>>(
        curr, Wq_t, q, 64, 32ll * 64 * 64, 64, M_q);
    gemm64_wmma<float, __bf16, false><<<(int)(M_q / 128), 32, 0, stream>>>(
        curr, Wq_s, q2, 64, 32ll * 64 * 64, 64, M_q);

    // 2) stage-1 attention: (b,h,j-group) blocks
    stage1_attn<<<128 * 2 * 16, 32, 0, stream>>>(q, kbuf, vbuf, Tpre);

    // 3) T = Tpre @ Wo_t^T
    gemm64_wmma<__bf16, __bf16, true><<<(int)(M_T / 128), 32, smem_bf16, stream>>>(
        Tpre, Wo_t, Tm, M_T, 0, 64, M_T);

    // 4) k2 = T @ Wk_s^T (region A), v2 = T @ Wv_s^T (region B)
    gemm64_wmma<__bf16, __bf16, true><<<(int)(M_T / 128), 32, smem_bf16, stream>>>(
        Tm, Wk_s, k2, M_T, 0, 64, M_T);
    gemm64_wmma<__bf16, __bf16, true><<<(int)(M_T / 128), 32, smem_bf16, stream>>>(
        Tm, Wv_s, v2, M_T, 0, 64, M_T);

    // 5) stage-2 attention (GEMV-shaped)
    stage2_attn<<<128 * 64, 64, 0, stream>>>(q2, k2, v2, adj, opre);

    // 6) out = opre @ Wo_s^T (f32 output)
    gemm64_wmma<__bf16, float, true><<<(int)(M_q / 128), 32, smem_bf16, stream>>>(
        opre, Wo_s, out, M_q, 0, 64, M_q);
}
